// SimpleGNNLayer_69209103008309
// MI455X (gfx1250) — matrix operand order repaired and checked by reading, not compiled, not run on hardware
//
#include <hip/hip_runtime.h>

// ---- problem constants (from reference) ----
#define N_NODES 50000
#define N_EDGES 800000
#define D 128          // D_IN == D_OUT == 128

typedef __attribute__((ext_vector_type(16))) __bf16 v16bf;
typedef __attribute__((ext_vector_type(8)))  __bf16 v8bf;
typedef __attribute__((ext_vector_type(8)))  float  v8f;

// ---------------------------------------------------------------------------
// Kernel 1: zero agg + deg, convert W (f32 [128x128] row-major) -> bf16
// ---------------------------------------------------------------------------
__global__ void gnn_init_kernel(float4* __restrict__ agg4, int nAgg4,
                                float* __restrict__ deg, int nDeg,
                                __bf16* __restrict__ wb,
                                const float* __restrict__ W, int nW) {
    int i = blockIdx.x * blockDim.x + threadIdx.x;
    if (i < nAgg4) {
        agg4[i] = make_float4(0.f, 0.f, 0.f, 0.f);
        return;
    }
    int j = i - nAgg4;
    if (j < nDeg) {
        deg[j] = 0.f;
        return;
    }
    int k = j - nDeg;
    if (k < nW) {
        wb[k] = (__bf16)W[k];
    }
}

// ---------------------------------------------------------------------------
// Kernel 2: scatter-add, one wave32 per edge.
//   agg[row][:] += x[col][:]   (4 atomics per lane, coalesced 128B row reads)
//   deg[row]    += 1           (lane 0)
// x (25.6MB) and agg (25.6MB) both fit in the 192MB L2, so the random gathers
// and atomics resolve at L2, not HBM.
// ---------------------------------------------------------------------------
__global__ void gnn_scatter_kernel(const float* __restrict__ x,
                                   const long long* __restrict__ ei,
                                   float* __restrict__ agg,
                                   float* __restrict__ deg) {
    int wave = (blockIdx.x * blockDim.x + threadIdx.x) >> 5;
    int lane = threadIdx.x & 31;
    if (wave >= N_EDGES) return;
    int row = (int)ei[wave];            // edge_index[0][e]
    int col = (int)ei[N_EDGES + wave];  // edge_index[1][e]
    const float* src = x + (long long)col * D;
    float*       dst = agg + (long long)row * D;
#pragma unroll
    for (int t = 0; t < 4; ++t) {
        int d = lane + 32 * t;
        unsafeAtomicAdd(dst + d, src[d]);   // global_atomic_add_f32
    }
    if (lane == 0) unsafeAtomicAdd(deg + row, 1.0f);
}

// ---------------------------------------------------------------------------
// Kernel 3: y = relu((agg/deg) @ W^T + b) via v_wmma_f32_16x16x32_bf16.
// One block = 16 output rows; 8 waves each own a 16-col tile; K=128 -> 4 WMMA.
// LDS stages the normalized bf16 A-tile with padded stride (bank-conflict free).
// ---------------------------------------------------------------------------
#define LDST 136   // 128 + 8 bf16 pad -> 272B row stride, conflict-free

__global__ void gnn_gemm_kernel(const float* __restrict__ agg,
                                const float* __restrict__ deg,
                                const __bf16* __restrict__ wb,
                                const float* __restrict__ bias,
                                float* __restrict__ out) {
    __shared__ __bf16 sA[16 * LDST];
    const int m0  = blockIdx.x * 16;
    const int tid = threadIdx.x;

    // ---- stage: normalize 16x128 f32 tile -> bf16 in LDS ----
    {
        int r   = tid >> 4;          // 0..15 row in tile
        int c8  = (tid & 15) * 8;    // 8-col chunk
        int gR  = m0 + r;
        float rd = 1.0f / fmaxf(deg[gR], 1.0f);
        const float4* p = (const float4*)(agg + (long long)gR * D + c8);
        float4 f0 = p[0], f1 = p[1];
        v8bf h;
        h[0] = (__bf16)(f0.x * rd); h[1] = (__bf16)(f0.y * rd);
        h[2] = (__bf16)(f0.z * rd); h[3] = (__bf16)(f0.w * rd);
        h[4] = (__bf16)(f1.x * rd); h[5] = (__bf16)(f1.y * rd);
        h[6] = (__bf16)(f1.z * rd); h[7] = (__bf16)(f1.w * rd);
        *(v8bf*)(sA + r * LDST + c8) = h;     // ds_store_b128
    }
    __syncthreads();

    const int lane  = tid & 31;
    const int wave  = tid >> 5;      // 0..7
    const int n0    = wave * 16;
    const int half  = lane >> 4;     // K-half select (this code's contiguous k order)
    const int mrow  = lane & 15;     // A: row per lane
    const int ncol  = lane & 15;     // B/C/D: col per lane

    v8f acc = {};
#pragma unroll
    for (int kb = 0; kb < 4; ++kb) {
        const int k0 = kb * 32;
        // A fragment (16x32 bf16): lane holds row mrow, K = half*16 + j (j = 0..15)
        v8bf a0 = *(const v8bf*)(sA + mrow * LDST + k0 + half * 16);
        v8bf a1 = *(const v8bf*)(sA + mrow * LDST + k0 + half * 16 + 8);
        v16bf A = __builtin_shufflevector(a0, a1,
            0,1,2,3,4,5,6,7,8,9,10,11,12,13,14,15);
        // B fragment (32x16 bf16): lane holds column ncol, K = half*16 + j.
        // B[k][n] = W^T[k][n] = W[n][k] -> contiguous 32B from wb row ncol.
        const v8bf* bp = (const v8bf*)(wb + (long long)(n0 + ncol) * D + k0 + half * 16);
        v16bf B = __builtin_shufflevector(bp[0], bp[1],
            0,1,2,3,4,5,6,7,8,9,10,11,12,13,14,15);
        acc = __builtin_amdgcn_wmma_f32_16x16x32_bf16(
            /*neg_a=*/false, A, /*neg_b=*/false, B,
            /*c_mod=*/(short)0, acc, /*reuse_a=*/false, /*reuse_b=*/false);
    }

    // ---- epilogue: bias + relu; C/D layout: VGPR v -> row half*8+v, col ncol ----
    float bn = bias[n0 + ncol];
    int mbase = m0 + half * 8;
#pragma unroll
    for (int v = 0; v < 8; ++v) {
        out[(long long)(mbase + v) * D + n0 + ncol] = fmaxf(acc[v] + bn, 0.0f);
    }
}

// ---------------------------------------------------------------------------
extern "C" void kernel_launch(void* const* d_in, const int* in_sizes, int n_in,
                              void* d_out, int out_size, void* d_ws, size_t ws_size,
                              hipStream_t stream) {
    const float*     x    = (const float*)d_in[0];
    const long long* ei   = (const long long*)d_in[1];   // int64 [2, N_EDGES]
    const float*     W    = (const float*)d_in[2];       // [128,128] row-major
    const float*     bias = (const float*)d_in[3];       // [128]
    float*           out  = (float*)d_out;               // [50000,128]

    // workspace layout: agg f32 [N,128] | deg f32 [N] | W bf16 [128,128]
    char*   ws  = (char*)d_ws;
    float*  agg = (float*)ws;
    float*  deg = (float*)(ws + (size_t)N_NODES * D * sizeof(float));
    __bf16* wb  = (__bf16*)(ws + (size_t)N_NODES * D * sizeof(float)
                               + (size_t)N_NODES * sizeof(float));

    // 1) init: zero agg (as float4) + deg, convert W -> bf16
    const int nAgg4 = N_NODES * D / 4;     // 1,600,000
    const int nTot  = nAgg4 + N_NODES + D * D;
    gnn_init_kernel<<<(nTot + 255) / 256, 256, 0, stream>>>(
        (float4*)agg, nAgg4, deg, N_NODES, wb, W, D * D);

    // 2) scatter: one wave per edge (8 waves / 256-thread block)
    gnn_scatter_kernel<<<N_EDGES / 8, 256, 0, stream>>>(x, ei, agg, deg);

    // 3) normalize + GEMM(WMMA bf16) + bias + relu; 50000/16 = 3125 blocks
    gnn_gemm_kernel<<<N_NODES / 16, 256, 0, stream>>>(agg, deg, wb, bias, out);
}
